// ClusteringMerger_82403242541299
// MI455X (gfx1250) — matrix operand order repaired and checked by reading, not compiled, not run on hardware
//
#include <hip/hip_runtime.h>
#include <hip/hip_bf16.h>

// CDNA5 / gfx1250: wave32, WMMA 16x16x32 bf16 + async LDS staging.
typedef __attribute__((ext_vector_type(16))) __bf16 v16bf;
typedef __attribute__((ext_vector_type(8)))  float  v8f;
typedef __attribute__((ext_vector_type(4)))  int    v4i;
// Address-space-qualified pointers for the gfx1250 async-LDS builtin:
typedef __attribute__((address_space(1))) v4i* gas_v4i;   // global
typedef __attribute__((address_space(3))) v4i* las_v4i;   // LDS

#define EPSN 1e-12f
#define FBIG 3.4e38f
#define BSTRIDE 136   // 128 elems + 8 pad -> 272B row stride, rotates LDS banks

#if defined(__has_builtin)
#if __has_builtin(__builtin_amdgcn_global_load_async_to_lds_b128)
#define HAVE_ASYNC_LDS 1
#endif
#endif
#ifndef HAVE_ASYNC_LDS
#define HAVE_ASYNC_LDS 0
#endif

__device__ __forceinline__ unsigned short f2bf(float f) {
  // round-to-nearest-even fp32 -> bf16
  unsigned u = __float_as_uint(f);
  unsigned r = 0x7FFFu + ((u >> 16) & 1u);
  return (unsigned short)((u + r) >> 16);
}

__device__ __forceinline__ float waveReduceSum(float v) {
  #pragma unroll
  for (int m = 16; m >= 1; m >>= 1) v += __shfl_xor(v, m, 32);
  return v;
}

// Stage one 16B segment global->LDS (async on gfx1250: tracked by ASYNCcnt).
__device__ __forceinline__ void stage16(const unsigned short* gsrc,
                                        unsigned short* ldst) {
#if HAVE_ASYNC_LDS
  __builtin_amdgcn_global_load_async_to_lds_b128(
      (gas_v4i)gsrc, (las_v4i)ldst, 0, 0);
#else
  uint4 t = *(const uint4*)gsrc;
  *(uint4*)ldst = t;
#endif
}

__device__ __forceinline__ void wait_stage() {
#if HAVE_ASYNC_LDS
#if __has_builtin(__builtin_amdgcn_s_wait_asynccnt)
  __builtin_amdgcn_s_wait_asynccnt(0);
#else
  asm volatile("s_wait_asynccnt 0x0" ::: "memory");
#endif
#endif
}

// prm layout: [0]=r0 [1]=r1 [2]=P [3]=K [4]=stride [5]=maxtok
__global__ void params_kernel(const long long* __restrict__ pr,
                              const int* __restrict__ mergep,
                              int* __restrict__ prm, int S) {
  int r0 = (int)pr[0];
  int r1 = (int)pr[1];
  int P = r1 - r0 + 1;
  int merge = mergep[0];
  if (merge < 1) merge = 1;
  int K = P / merge; if (K < 1) K = 1;
  prm[0] = r0;
  prm[1] = r1;
  prm[2] = P;
  prm[3] = K;
  prm[4] = P / K;                       // init stride
  prm[5] = (S + merge - 1) / merge;     // max_tokens
}

// L2-normalize patch rows; write fp32 + bf16 copies. One wave per row.
__global__ __launch_bounds__(256) void normalize_kernel(
    const float* __restrict__ hs, float* __restrict__ xn,
    unsigned short* __restrict__ xnbf, const int* __restrict__ prm,
    int S, int D) {
  const int r0 = prm[0], P = prm[2];
  const int b = blockIdx.y;
  const int p = blockIdx.x * 8 + (threadIdx.x >> 5);
  if (p >= P) return;
  const int lane = threadIdx.x & 31;
  const float* src = hs + ((size_t)b * S + r0 + p) * D;
  float ss = 0.f;
  for (int d = lane * 4; d < D; d += 128) {
    float4 v = *(const float4*)(src + d);
    ss += v.x * v.x + v.y * v.y + v.z * v.z + v.w * v.w;
  }
  ss = waveReduceSum(ss);
  const float inv = 1.0f / fmaxf(sqrtf(ss), EPSN);
  float* dst = xn + ((size_t)b * S + p) * D;
  unsigned short* dbf = xnbf + ((size_t)b * S + p) * D;
  for (int d = lane * 4; d < D; d += 128) {
    float4 v = *(const float4*)(src + d);
    v.x *= inv; v.y *= inv; v.z *= inv; v.w *= inv;
    *(float4*)(dst + d) = v;
    dbf[d] = f2bf(v.x); dbf[d + 1] = f2bf(v.y);
    dbf[d + 2] = f2bf(v.z); dbf[d + 3] = f2bf(v.w);
  }
}

// Evenly-spaced init: centroid k = xn[k*stride]. One wave per centroid.
__global__ __launch_bounds__(256) void init_cent_kernel(
    const float* __restrict__ xn, float* __restrict__ cent,
    unsigned short* __restrict__ centbf, float* __restrict__ cnorm,
    const int* __restrict__ prm, int S, int D) {
  const int K = prm[3], stride = prm[4];
  const int b = blockIdx.y;
  const int k = blockIdx.x * 8 + (threadIdx.x >> 5);
  if (k >= K) return;
  const int lane = threadIdx.x & 31;
  const float* src = xn + ((size_t)b * S + (size_t)k * stride) * D;
  float* dst = cent + ((size_t)b * S + k) * D;
  unsigned short* dbf = centbf + ((size_t)b * S + k) * D;
  float ss = 0.f;
  for (int d = lane * 4; d < D; d += 128) {
    float4 v = *(const float4*)(src + d);
    *(float4*)(dst + d) = v;
    dbf[d] = f2bf(v.x); dbf[d + 1] = f2bf(v.y);
    dbf[d + 2] = f2bf(v.z); dbf[d + 3] = f2bf(v.w);
    ss += v.x * v.x + v.y * v.y + v.z * v.z + v.w * v.w;
  }
  ss = waveReduceSum(ss);
  if (lane == 0) cnorm[(size_t)b * S + k] = ss;
}

// WMMA assignment: argmin_k (||c_k||^2 - 2 * xn . c_k)   (x2==1 is constant).
// Block = 8 waves = 128 rows; all waves sweep the SAME 16-centroid column
// tile, staged into double-buffered LDS with async global->LDS copies.
// Assumes D == 128 (the workload).
__global__ __launch_bounds__(256) void assign_kernel(
    const unsigned short* __restrict__ xnbf,
    const unsigned short* __restrict__ centbf,
    const float* __restrict__ cnorm,
    int* __restrict__ labels,
    const int* __restrict__ prm, int S, int D) {
  const int P = prm[2], K = prm[3];
  const int b = blockIdx.y;
  const int tid = threadIdx.x;
  const int lane = tid & 31;
  const int wv = tid >> 5;
  const int half = lane >> 4;   // wave32 halves
  const int m16 = lane & 15;
  const int rowbase = (blockIdx.x * 8 + wv) * 16;
  const size_t cb = (size_t)b * S;

  __shared__ unsigned short btile[2][16 * BSTRIDE];

  // ---- A fragments: 16x32 bf16 per chunk, 4 chunks (D=128), per ISA layout:
  // lane L (row M=L&15): VGPR v packs K = 2*(v&3) + 16*(v>>2) + 8*half
  int arow = rowbase + m16; if (arow > P - 1) arow = P - 1;
  if (arow < 0) arow = 0;
  const unsigned short* rp = xnbf + (cb + arow) * (size_t)D;
  union AF { v16bf v; unsigned u[8]; };
  AF a[4];
  #pragma unroll
  for (int c = 0; c < 4; ++c) {
    #pragma unroll
    for (int v = 0; v < 8; ++v) {
      const int k = ((v & 3) * 2) + ((v >> 2) * 16) + half * 8;
      a[c].u[v] = *(const unsigned*)(rp + c * 32 + k);
    }
  }

  float bd[8]; int bi[8];
  #pragma unroll
  for (int r = 0; r < 8; ++r) { bd[r] = FBIG; bi[r] = 0; }

  const int numCT = (K + 15) >> 4;
  const int trow = tid >> 4;        // 0..15: centroid row this thread stages
  const int tseg = (tid & 15) * 8;  // 16B segment within the row

  // prologue: stage tile 0 into buffer 0
  {
    int n = trow; if (n > K - 1) n = K - 1;
    stage16(centbf + (cb + n) * (size_t)D + tseg,
            &btile[0][trow * BSTRIDE + tseg]);
  }

  int cur = 0;
  for (int ct = 0; ct < numCT; ++ct) {
    wait_stage();       // our async stage for btile[cur] has landed
    __syncthreads();    // ...and everyone is done reading btile[cur^1]

    if (ct + 1 < numCT) {
      int n = (ct + 1) * 16 + trow; if (n > K - 1) n = K - 1;
      stage16(centbf + (cb + n) * (size_t)D + tseg,
              &btile[cur ^ 1][trow * BSTRIDE + tseg]);
    }
    if (ct + 2 < numCT)  // warm L2 for the tile after next (global_prefetch_b8)
      __builtin_prefetch(centbf + (cb + (size_t)(ct + 2) * 16 + trow) * (size_t)D + tseg, 0, 1);

    const int colbase = ct << 4;
    const int col = colbase + m16;
    const float cn = (col < K) ? cnorm[cb + col] : FBIG;

    // B fragments 32x16 bf16: lane = column N = m16, K = 8*half+{0..7} and
    // 16+8*half+{0..7}, read from the padded LDS tile (conflict-free).
    const unsigned short* cp = &btile[cur][m16 * BSTRIDE + half * 8];
    v8f acc = {};
    #pragma unroll
    for (int c = 0; c < 4; ++c) {
      union BU { v16bf v; uint4 q[2]; } bu;
      bu.q[0] = *(const uint4*)(cp + c * 32);
      bu.q[1] = *(const uint4*)(cp + c * 32 + 16);
      acc = __builtin_amdgcn_wmma_f32_16x16x32_bf16(
          false, a[c].v, false, bu.v, (short)0, acc, false, false);
    }
    // C/D layout: VGPR r of lane L -> (M = r + 8*(L>>4), N = L&15).
    // Per-lane columns arrive in strictly increasing index order, so plain
    // strict '<' already keeps the lowest-index minimum within a lane; the
    // equality tie-break is only needed once in the cross-lane butterfly.
    #pragma unroll
    for (int r = 0; r < 8; ++r) {
      const float d = cn - 2.0f * acc[r];
      if (d < bd[r]) { bd[r] = d; bi[r] = col; }
    }
    cur ^= 1;
  }

  // argmin across the 16 N-lanes (butterfly; masks < 16 stay in each half);
  // exact tie-break to the lowest index, matching jnp.argmin semantics.
  #pragma unroll
  for (int msk = 1; msk < 16; msk <<= 1) {
    #pragma unroll
    for (int r = 0; r < 8; ++r) {
      const float od = __shfl_xor(bd[r], msk, 32);
      const int   oi = __shfl_xor(bi[r], msk, 32);
      if (od < bd[r] || (od == bd[r] && oi < bi[r])) { bd[r] = od; bi[r] = oi; }
    }
  }
  if (m16 == 0) {
    #pragma unroll
    for (int r = 0; r < 8; ++r) {
      const int row = rowbase + r + 8 * half;
      if (row < P) labels[cb + row] = bi[r];
    }
  }
}

// Segment sums via L2 atomics. One wave per point.
__global__ __launch_bounds__(256) void update_kernel(
    const float* __restrict__ xn, const int* __restrict__ labels,
    float* __restrict__ sums, float* __restrict__ counts,
    const int* __restrict__ prm, int S, int D) {
  const int P = prm[2];
  const int b = blockIdx.y;
  const int p = blockIdx.x * 8 + (threadIdx.x >> 5);
  if (p >= P) return;
  const int lane = threadIdx.x & 31;
  const int lab = labels[(size_t)b * S + p];
  const float* src = xn + ((size_t)b * S + p) * D;
  float* dst = sums + ((size_t)b * S + lab) * D;
  for (int d = lane * 4; d < D; d += 128) {
    float4 v = *(const float4*)(src + d);
    atomicAdd(dst + d, v.x);
    atomicAdd(dst + d + 1, v.y);
    atomicAdd(dst + d + 2, v.z);
    atomicAdd(dst + d + 3, v.w);
  }
  if (lane == 0) atomicAdd(counts + (size_t)b * S + lab, 1.0f);
}

// c = cnt>0 ? sums/cnt : c ; refresh bf16 copy and ||c||^2. One wave per k.
__global__ __launch_bounds__(256) void finalize_kernel(
    const float* __restrict__ sums, const float* __restrict__ counts,
    float* __restrict__ cent, unsigned short* __restrict__ centbf,
    float* __restrict__ cnorm, const int* __restrict__ prm, int S, int D) {
  const int K = prm[3];
  const int b = blockIdx.y;
  const int k = blockIdx.x * 8 + (threadIdx.x >> 5);
  if (k >= K) return;
  const int lane = threadIdx.x & 31;
  const size_t base = ((size_t)b * S + k) * (size_t)D;
  const float cnt = counts[(size_t)b * S + k];
  const float inv = (cnt > 0.f) ? (1.0f / cnt) : 0.f;
  float ss = 0.f;
  for (int d = lane * 4; d < D; d += 128) {
    float4 v;
    if (cnt > 0.f) {
      v = *(const float4*)(sums + base + d);
      v.x *= inv; v.y *= inv; v.z *= inv; v.w *= inv;
      *(float4*)(cent + base + d) = v;
    } else {
      v = *(const float4*)(cent + base + d);
    }
    centbf[base + d] = f2bf(v.x); centbf[base + d + 1] = f2bf(v.y);
    centbf[base + d + 2] = f2bf(v.z); centbf[base + d + 3] = f2bf(v.w);
    ss += v.x * v.x + v.y * v.y + v.z * v.z + v.w * v.w;
  }
  ss = waveReduceSum(ss);
  if (lane == 0) cnorm[(size_t)b * S + k] = ss;
}

// Assemble output: zeros | pooled(l2-normalized means, masked) | tail copy,
// plus the attention mask appended after the hidden block.
__global__ __launch_bounds__(128) void output_kernel(
    const float* __restrict__ hs, const float* __restrict__ sums,
    const float* __restrict__ counts, float* __restrict__ out,
    const int* __restrict__ prm, int S, int D, int T) {
  const int r1 = prm[1], K = prm[3], maxtok = prm[5];
  const int b = blockIdx.y;
  const int t = blockIdx.x;
  if (t >= T) return;
  __shared__ float red[128];
  const int tidx = threadIdx.x;
  float* orow = out + ((size_t)b * T + t) * D;
  float mval;
  if (t < maxtok - K) {
    for (int d = tidx; d < D; d += 128) orow[d] = 0.f;
    mval = 0.f;
  } else if (t < maxtok) {
    const int k = t - (maxtok - K);
    const float cnt = counts[(size_t)b * S + k];
    const float inv = 1.0f / fmaxf(cnt, 1.0f);
    const size_t base = ((size_t)b * S + k) * (size_t)D;
    float ss = 0.f;
    for (int d = tidx; d < D; d += 128) {
      const float mn = sums[base + d] * inv;
      ss += mn * mn;
    }
    red[tidx] = ss;
    __syncthreads();
    #pragma unroll
    for (int off = 64; off >= 1; off >>= 1) {
      if (tidx < off) red[tidx] += red[tidx + off];
      __syncthreads();
    }
    const float nrm = sqrtf(red[0]);
    const float scale = (cnt > 0.f) ? (inv / fmaxf(nrm, EPSN)) : 0.f;
    for (int d = tidx; d < D; d += 128) orow[d] = sums[base + d] * scale;
    mval = (cnt > 0.f) ? 1.f : 0.f;
  } else {
    const int srcrow = r1 + 2 + (t - maxtok);
    const float* src = hs + ((size_t)b * S + srcrow) * D;
    for (int d = tidx; d < D; d += 128) orow[d] = src[d];
    mval = 1.f;
  }
  if (tidx == 0) out[(size_t)gridDim.y * T * D + (size_t)b * T + t] = mval;
}

extern "C" void kernel_launch(void* const* d_in, const int* in_sizes, int n_in,
                              void* d_out, int out_size, void* d_ws, size_t ws_size,
                              hipStream_t stream) {
  (void)n_in; (void)ws_size;
  const float*     hs     = (const float*)d_in[0];
  const long long* pr     = (const long long*)d_in[2];
  const int*       mergep = (const int*)d_in[3];
  const int B = in_sizes[2] / 2;
  const int S = in_sizes[1] / B;
  const int D = in_sizes[0] / in_sizes[1];
  float* out = (float*)d_out;
  const int T = out_size / (B * (D + 1));   // max_tokens + tail_len

  // Workspace layout (capacities use S as worst case for P and K).
  char* ws = (char*)d_ws;
  size_t off = 0;
  auto wsalloc = [&](size_t bytes) -> char* {
    char* p = ws + off;
    off = (off + bytes + 255) & ~(size_t)255;
    return p;
  };
  int*            prm    = (int*)wsalloc(256);
  float*          xn     = (float*)wsalloc((size_t)B * S * D * 4);
  unsigned short* xnbf   = (unsigned short*)wsalloc((size_t)B * S * D * 2);
  float*          cent   = (float*)wsalloc((size_t)B * S * D * 4);
  unsigned short* centbf = (unsigned short*)wsalloc((size_t)B * S * D * 2);
  float*          cnorm  = (float*)wsalloc((size_t)B * S * 4);
  int*            labels = (int*)wsalloc((size_t)B * S * 4);
  float*          sums   = (float*)wsalloc((size_t)B * S * D * 4);
  float*          counts = (float*)wsalloc((size_t)B * S * 4);

  params_kernel<<<1, 1, 0, stream>>>(pr, mergep, prm, S);

  const dim3 blk(256);
  const dim3 rows8((S + 7) / 8, B);
  const dim3 tiles((S + 127) / 128, B);   // 8 row-tiles (128 rows) per block

  normalize_kernel<<<rows8, blk, 0, stream>>>(hs, xn, xnbf, prm, S, D);
  init_cent_kernel<<<rows8, blk, 0, stream>>>(xn, cent, centbf, cnorm, prm, S, D);

  for (int it = 0; it < 10; ++it) {
    assign_kernel<<<tiles, blk, 0, stream>>>(xnbf, centbf, cnorm, labels, prm, S, D);
    (void)hipMemsetAsync(sums, 0, (size_t)B * S * D * 4, stream);
    (void)hipMemsetAsync(counts, 0, (size_t)B * S * 4, stream);
    update_kernel<<<rows8, blk, 0, stream>>>(xn, labels, sums, counts, prm, S, D);
    finalize_kernel<<<rows8, blk, 0, stream>>>(sums, counts, cent, centbf, cnorm, prm, S, D);
  }
  // Final assignment + pooled statistics
  assign_kernel<<<tiles, blk, 0, stream>>>(xnbf, centbf, cnorm, labels, prm, S, D);
  (void)hipMemsetAsync(sums, 0, (size_t)B * S * D * 4, stream);
  (void)hipMemsetAsync(counts, 0, (size_t)B * S * 4, stream);
  update_kernel<<<rows8, blk, 0, stream>>>(xn, labels, sums, counts, prm, S, D);

  const dim3 og((unsigned)T, (unsigned)B);
  output_kernel<<<og, dim3(128), 0, stream>>>(hs, sums, counts, out, prm, S, D, T);
}
